// FraudGAT_63556926046213
// MI455X (gfx1250) — matrix operand order repaired and checked by reading, not compiled, not run on hardware
//
#include <hip/hip_runtime.h>
#include <hip/hip_bf16.h>

typedef __attribute__((ext_vector_type(16))) _Float16 v16h;
typedef __attribute__((ext_vector_type(8)))  _Float16 v8h;
typedef __attribute__((ext_vector_type(4)))  _Float16 v4h;
typedef __attribute__((ext_vector_type(8)))  float    v8f;

#define N_TX 100000
#define N_ACC 50000
#define N_MER 5000
#define HIDD 128
#define HCW  512
#define NEDGE 100000

// ---------------------------------------------------------------- fill
__global__ void fill_kernel(float* __restrict__ p, long n, float v) {
  long i = (long)blockIdx.x * blockDim.x + threadIdx.x;
  long st = (long)gridDim.x * blockDim.x;
  for (; i < n; i += st) p[i] = v;
}

// ---------------------------------------------------------------- WMMA GEMM
// C[M,N] = act(A[M,K] @ B[K,N] + bias). Requires K%32==0; N handled with masking
// in tiles of 128. Block tile 128x128, 8 waves (4m x 2n), wave tile 32x64
// = 2x4 wmma 16x16x32 frags -> 8 v_wmma per k-step vs 12 b128 LDS frag loads.
#define LDH 40  // padded LDS row stride in halfs (80B: 16B aligned, conflict-free)
__global__ __launch_bounds__(256) void gemm_wmma_kernel(
    const float* __restrict__ A, const float* __restrict__ B,
    const float* __restrict__ bias, float* __restrict__ C,
    int M, int N, int K, int relu)
{
  __shared__ __align__(16) _Float16 As[128 * LDH];
  __shared__ __align__(16) _Float16 Bs[128 * LDH];
  const int tid  = threadIdx.x;
  const int m0   = blockIdx.y * 128;
  const int n0   = blockIdx.x * 128;
  const int wave = tid >> 5, lane = tid & 31;
  const int wm = wave >> 1, wn = wave & 1;
  const int l  = lane & 15;
  const int hi = (lane >> 4) & 1;
  const bool mfull = (m0 + 128 <= M);

  v8f z = {0.f,0.f,0.f,0.f,0.f,0.f,0.f,0.f};
  v8f acc[2][4];
  #pragma unroll
  for (int mi = 0; mi < 2; ++mi)
    #pragma unroll
    for (int ni = 0; ni < 4; ++ni) acc[mi][ni] = z;

  // staging coordinates
  const int ar  = tid >> 3;          // A: row 0..31 (+32 per pass)
  const int akq = (tid & 7) * 4;     // A: k quad 0,4,..28
  const int bn    = tid & 127;       // B: tile col 0..127
  const int bkb   = (tid >> 7) * 16; // B: k base 0 or 16
  const int gn    = n0 + bn;
  const bool nok  = (gn < N);

  for (int k0 = 0; k0 < K; k0 += 32) {
    // ---- stage A tile 128x32 (fp32 -> f16)
    if (mfull) {
      #pragma unroll
      for (int pass = 0; pass < 4; ++pass) {
        int r = ar + pass * 32;
        float4 v4 = *(const float4*)(A + (long)(m0 + r) * K + k0 + akq);
        v4h hv = {(_Float16)v4.x, (_Float16)v4.y, (_Float16)v4.z, (_Float16)v4.w};
        *(v4h*)(As + r * LDH + akq) = hv;
      }
    } else {
      #pragma unroll
      for (int pass = 0; pass < 4; ++pass) {
        int r = ar + pass * 32;
        float4 v4 = make_float4(0.f, 0.f, 0.f, 0.f);
        if (m0 + r < M) v4 = *(const float4*)(A + (long)(m0 + r) * K + k0 + akq);
        v4h hv = {(_Float16)v4.x, (_Float16)v4.y, (_Float16)v4.z, (_Float16)v4.w};
        *(v4h*)(As + r * LDH + akq) = hv;
      }
    }
    // ---- stage B tile 32x128 transposed into Bs[n][k] (4x ds_store_b64)
    #pragma unroll
    for (int g = 0; g < 4; ++g) {
      v4h hv;
      #pragma unroll
      for (int kk = 0; kk < 4; ++kk) {
        int k = bkb + g * 4 + kk;
        float v = nok ? B[(long)(k0 + k) * N + gn] : 0.f;
        hv[kk] = (_Float16)v;
      }
      *(v4h*)(Bs + bn * LDH + bkb + g * 4) = hv;
    }
    __syncthreads();

    v16h afrag[2], bfrag[4];
    #pragma unroll
    for (int mi = 0; mi < 2; ++mi) {
      // A 16x32 layout: lanes 0-15 -> K 0..7 & 16..23 ; lanes 16-31 -> K 8..15 & 24..31
      const _Float16* ab = As + (wm * 32 + mi * 16 + l) * LDH + hi * 8;
      v8h alo = *(const v8h*)(ab);
      v8h ahi = *(const v8h*)(ab + 16);
      afrag[mi] = __builtin_shufflevector(alo, ahi,
          0,1,2,3,4,5,6,7,8,9,10,11,12,13,14,15);
    }
    #pragma unroll
    for (int ni = 0; ni < 4; ++ni) {
      // B 32x16 layout: lanes 0-15 -> K 0..7 & 16..23 ; lanes 16-31 -> K 8..15 & 24..31 (col = l)
      const _Float16* bb = Bs + (wn * 64 + ni * 16 + l) * LDH + hi * 8;
      v8h blo = *(const v8h*)(bb);
      v8h bhi = *(const v8h*)(bb + 16);
      bfrag[ni] = __builtin_shufflevector(blo, bhi,
          0,1,2,3,4,5,6,7,8,9,10,11,12,13,14,15);
    }
    #pragma unroll
    for (int mi = 0; mi < 2; ++mi)
      #pragma unroll
      for (int ni = 0; ni < 4; ++ni)
        acc[mi][ni] = __builtin_amdgcn_wmma_f32_16x16x32_f16(
            false, afrag[mi], false, bfrag[ni], (short)0, acc[mi][ni], false, false);
    __syncthreads();
  }

  // epilogue: C/D layout -> lane holds col l, VGPR v holds row v (+8 for hi half)
  #pragma unroll
  for (int mi = 0; mi < 2; ++mi) {
    #pragma unroll
    for (int ni = 0; ni < 4; ++ni) {
      int col = n0 + wn * 64 + ni * 16 + l;
      if (col < N) {
        float bv = bias ? bias[col] : 0.f;
        int rb = m0 + wm * 32 + mi * 16 + hi * 8;
        #pragma unroll
        for (int v = 0; v < 8; ++v) {
          int row = rb + v;
          if (mfull || row < M) {
            float y = acc[mi][ni][v] + bv;
            if (relu) y = y > 0.f ? y : 0.f;
            C[(long)row * N + col] = y;
          }
        }
      }
    }
  }
}

// ---------------------------------------------------------------- attention prep
// wd[r][h][k] = sum_c W_r[k, h*128+c] * att_d[r][h][c]
__global__ void wd_kernel(const float* __restrict__ W, const float* __restrict__ attd,
                          float* __restrict__ wd, int K) {
  int id = blockIdx.x * blockDim.x + threadIdx.x;
  if (id >= 6 * 4 * K) return;
  int k = id % K;
  int h = (id / K) & 3;
  int r = id / (4 * K);
  const float* Wr = W + (long)r * K * HCW;
  const float* a  = attd + (r * 4 + h) * 128;
  float s = 0.f;
  for (int c = 0; c < 128; ++c) s += Wr[(long)k * HCW + h * 128 + c] * a[c];
  wd[id] = s;
}

// ss[n,h] = dot(hs[n, h*128 : h*128+128], a_s[h])  (wave per (n,h))
__global__ void ss_kernel(const float* __restrict__ hs, const float* __restrict__ a_s,
                          float* __restrict__ ss, int n) {
  int gid = blockIdx.x * blockDim.x + threadIdx.x;
  int w = gid >> 5, lane = gid & 31;
  int node = w >> 2, h = w & 3;
  if (node >= n) return;
  const float* row = hs + (long)node * HCW + h * 128;
  const float* a   = a_s + h * 128;
  float s = 0.f;
  #pragma unroll
  for (int it = 0; it < 4; ++it) { int c = it * 32 + lane; s += row[c] * a[c]; }
  #pragma unroll
  for (int off = 16; off > 0; off >>= 1) s += __shfl_xor(s, off, 32);
  if (lane == 0) ss[node * 4 + h] = s;
}

// sd[n,h] = dot(hdst[n,:K], wd[h,:K])  (wave per (n,h))
__global__ void sd_kernel(const float* __restrict__ hdst, const float* __restrict__ wd,
                          float* __restrict__ sd, int n, int K) {
  int gid = blockIdx.x * blockDim.x + threadIdx.x;
  int w = gid >> 5, lane = gid & 31;
  int node = w >> 2, h = w & 3;
  if (node >= n) return;
  const float* row = hdst + (long)node * K;
  const float* wv  = wd + h * K;
  float s = 0.f;
  for (int c = lane; c < K; c += 32) s += row[c] * wv[c];
  #pragma unroll
  for (int off = 16; off > 0; off >>= 1) s += __shfl_xor(s, off, 32);
  if (lane == 0) sd[node * 4 + h] = s;
}

// ---------------------------------------------------------------- edge softmax
__device__ __forceinline__ unsigned fkey(float f) {
  unsigned u = __float_as_uint(f);
  return (u & 0x80000000u) ? ~u : (u | 0x80000000u);
}
__device__ __forceinline__ float funkey(unsigned k) {
  return __uint_as_float((k & 0x80000000u) ? (k & 0x7fffffffu) : ~k);
}

__global__ void edge_pass1(const int* __restrict__ src, const int* __restrict__ dst,
                           const float* __restrict__ ss, const float* __restrict__ sd,
                           float* __restrict__ ebuf, unsigned* __restrict__ m, int E) {
  int id = blockIdx.x * blockDim.x + threadIdx.x;
  if (id >= E * 4) return;
  int e = id >> 2, h = id & 3;
  float v = ss[src[e] * 4 + h] + sd[dst[e] * 4 + h];
  v = v > 0.f ? v : 0.2f * v;              // leaky_relu(0.2)
  ebuf[id] = v;
  atomicMax(&m[dst[e] * 4 + h], fkey(v));  // float segment-max via monotone key
}

__global__ void edge_pass2(const int* __restrict__ dst, float* __restrict__ ebuf,
                           const unsigned* __restrict__ m, float* __restrict__ denom, int E) {
  int id = blockIdx.x * blockDim.x + threadIdx.x;
  if (id >= E * 4) return;
  int e = id >> 2, h = id & 3;
  float mv = funkey(m[dst[e] * 4 + h]);
  float ex = __expf(ebuf[id] - mv);
  ebuf[id] = ex;
  atomicAdd(&denom[dst[e] * 4 + h], ex);
}

// wave per edge: out[dst] += alpha * hs[src] over 512 channels (head = c>>7)
__global__ void edge_scatter(const int* __restrict__ src, const int* __restrict__ dst,
                             const float* __restrict__ ebuf, const float* __restrict__ denom,
                             const float* __restrict__ hs, float* __restrict__ out, int E) {
  int gid = blockIdx.x * blockDim.x + threadIdx.x;
  int nw = (gridDim.x * blockDim.x) >> 5;
  int w = gid >> 5, lane = gid & 31;
  for (int e = w; e < E; e += nw) {
    int s = src[e], d = dst[e];
    float alpha[4];
    #pragma unroll
    for (int h = 0; h < 4; ++h)
      alpha[h] = ebuf[e * 4 + h] / (denom[d * 4 + h] + 1e-16f);
    const float* hrow = hs + (long)s * HCW;
    float* orow = out + (long)d * HCW;
    #pragma unroll
    for (int it = 0; it < 16; ++it) {
      int c = it * 32 + lane;
      atomicAdd(&orow[c], alpha[it >> 2] * hrow[c]);
    }
  }
}

// ---------------------------------------------------------------- batchnorm
__global__ void bn_stats(const float* __restrict__ x, float* __restrict__ stats, int nrows) {
  int r0 = blockIdx.x * 128;
  int rend = r0 + 128; if (rend > nrows) rend = nrows;
  for (int c = threadIdx.x; c < 512; c += 256) {
    float s = 0.f, s2 = 0.f;
    for (int r = r0; r < rend; ++r) {
      float v = x[(long)r * 512 + c];
      s += v; s2 += v * v;
    }
    atomicAdd(&stats[c], s);
    atomicAdd(&stats[512 + c], s2);
  }
}

__global__ void bn_finalize(const float* __restrict__ stats, const float* __restrict__ g,
                            const float* __restrict__ b, float* __restrict__ scale,
                            float* __restrict__ shift, float invn) {
  int c = blockIdx.x * blockDim.x + threadIdx.x;
  if (c >= 512) return;
  float mean = stats[c] * invn;
  float var  = stats[512 + c] * invn - mean * mean;
  float inv  = rsqrtf(var + 1e-5f);
  float sc   = g[c] * inv;
  scale[c] = sc;
  shift[c] = b[c] - mean * sc;
}

__global__ void apply_bn(float* __restrict__ t, const float* __restrict__ cur,
                         const float* __restrict__ scale, const float* __restrict__ shift,
                         long n, int res) {
  long i = (long)blockIdx.x * blockDim.x + threadIdx.x;
  if (i >= n) return;
  int c = (int)(i & 511);
  float y = t[i] * scale[c] + shift[c];
  y = y > 0.f ? y : 0.f;
  if (res) y += cur[i];
  t[i] = y;
}

__global__ void apply_plain(float* __restrict__ t, const float* __restrict__ cur,
                            const float* __restrict__ bsum, long n, int res) {
  long i = (long)blockIdx.x * blockDim.x + threadIdx.x;
  if (i >= n) return;
  int c = (int)(i & 511);
  float y = t[i] + bsum[c];
  y = y > 0.f ? y : 0.f;
  if (res) y += cur[i];
  t[i] = y;
}

__global__ void bsum_kernel(const float* __restrict__ attb, float* __restrict__ bacc,
                            float* __restrict__ bmer) {
  int c = blockIdx.x * blockDim.x + threadIdx.x;
  if (c >= 512) return;
  bacc[c] = attb[0 * 512 + c] + attb[1 * 512 + c];  // relations 0,1 -> acc
  bmer[c] = attb[2 * 512 + c];                      // relation 2   -> mer
}

// ---------------------------------------------------------------- head
__global__ void logits_kernel(const float* __restrict__ hidden, const float* __restrict__ w2,
                              const float* __restrict__ b2, float* __restrict__ out, int n) {
  int gid = blockIdx.x * blockDim.x + threadIdx.x;
  int w = gid >> 5, lane = gid & 31;
  if (w >= n) return;
  const float* row = hidden + (long)w * 64;
  float s = row[lane] * w2[lane] + row[lane + 32] * w2[lane + 32];
  #pragma unroll
  for (int off = 16; off > 0; off >>= 1) s += __shfl_xor(s, off, 32);
  if (lane == 0) out[w] = s + b2[0];
}

// ================================================================ launch
static inline int cdiv(long a, long b) { return (int)((a + b - 1) / b); }

extern "C" void kernel_launch(void* const* d_in, const int* in_sizes, int n_in,
                              void* d_out, int out_size, void* d_ws, size_t ws_size,
                              hipStream_t stream) {
  const float* x_tx  = (const float*)d_in[0];
  const float* x_acc = (const float*)d_in[1];
  const float* x_mer = (const float*)d_in[2];
  const float* Wtx   = (const float*)d_in[3];
  const float* btx   = (const float*)d_in[4];
  const float* Wacc  = (const float*)d_in[5];
  const float* bacc  = (const float*)d_in[6];
  const float* Wmer  = (const float*)d_in[7];
  const float* bmer  = (const float*)d_in[8];
  const float* W0    = (const float*)d_in[9];
  const float* Wrest = (const float*)d_in[10];
  const float* att_s = (const float*)d_in[11];
  const float* att_d = (const float*)d_in[12];
  const float* att_b = (const float*)d_in[13];
  const float* bn_g  = (const float*)d_in[14];
  const float* bn_b  = (const float*)d_in[15];
  const float* Wh1   = (const float*)d_in[16];
  const float* bh1   = (const float*)d_in[17];
  const float* Wh2   = (const float*)d_in[18];
  const float* bh2   = (const float*)d_in[19];
  const int*   edges = (const int*)d_in[20];

  const int n_of[3]  = {N_TX, N_ACC, N_MER};
  const int SRC_T[6] = {0, 0, 0, 1, 1, 2};
  const int DST_T[6] = {1, 1, 2, 0, 0, 0};

  float* ws = (float*)d_ws;
  long off = 0;
  auto alloc = [&](long n) { float* p = ws + off; off += n; return p; };
  float* htx[2]   = {alloc((long)N_TX * 512),  alloc((long)N_TX * 512)};
  float* haccb[2] = {alloc((long)N_ACC * 512), alloc((long)N_ACC * 512)};
  float* hmerb[2] = {alloc((long)N_MER * 512), alloc((long)N_MER * 512)};
  float* hs     = alloc((long)N_TX * 512);
  float* hidden = alloc((long)N_TX * 64);
  float* ssb    = alloc((long)N_TX * 4);
  float* sdb    = alloc((long)N_TX * 4);
  float* mb     = alloc((long)N_TX * 4);
  float* denb   = alloc((long)N_TX * 4);
  float* eb     = alloc((long)NEDGE * 4);
  float* wdb    = alloc(6L * 4 * 512);
  float* stats  = alloc(1024);
  float* scale  = alloc(512);
  float* shift  = alloc(512);
  float* bsacc  = alloc(512);
  float* bsmer  = alloc(512);

  // ---- input projections: h = relu(x @ W + b), width 128
  gemm_wmma_kernel<<<dim3(1, cdiv(N_TX, 128)), 256, 0, stream>>>(
      x_tx, Wtx, btx, htx[0], N_TX, HIDD, 64, 1);
  gemm_wmma_kernel<<<dim3(1, cdiv(N_ACC, 128)), 256, 0, stream>>>(
      x_acc, Wacc, bacc, haccb[0], N_ACC, HIDD, 32, 1);
  gemm_wmma_kernel<<<dim3(1, cdiv(N_MER, 128)), 256, 0, stream>>>(
      x_mer, Wmer, bmer, hmerb[0], N_MER, HIDD, 32, 1);

  int p = 0;
  for (int i = 0; i < 3; ++i) {
    const int dim = (i == 0) ? 128 : 512;
    float* cur[3] = {htx[p], haccb[p], hmerb[p]};
    float* acc[3] = {htx[1 - p], haccb[1 - p], hmerb[1 - p]};

    for (int t = 0; t < 3; ++t) {
      long n = (long)n_of[t] * 512;
      int blks = cdiv(n, 256); if (blks > 16384) blks = 16384;
      fill_kernel<<<blks, 256, 0, stream>>>(acc[t], n, 0.f);
    }

    const float* Wl = (i == 0) ? W0 : (Wrest + (long)(i - 1) * 6 * 512 * 512);
    wd_kernel<<<cdiv(6L * 4 * dim, 256), 256, 0, stream>>>(
        Wl, att_d + (long)i * 6 * 4 * 128, wdb, dim);

    for (int r = 0; r < 6; ++r) {
      const int st = SRC_T[r], dt = DST_T[r];
      const int ns = n_of[st], nd = n_of[dt];
      const float* Wr = Wl + (long)r * dim * 512;
      const int* srcp = edges + ((long)r * 2 + 0) * NEDGE;
      const int* dstp = edges + ((long)r * 2 + 1) * NEDGE;

      // hs = h_src @ W_r   [ns, 512]
      gemm_wmma_kernel<<<dim3(4, cdiv(ns, 128)), 256, 0, stream>>>(
          cur[st], Wr, nullptr, hs, ns, 512, dim, 0);
      // per-node attention scores
      ss_kernel<<<cdiv((long)ns * 128, 256), 256, 0, stream>>>(
          hs, att_s + ((long)i * 6 + r) * 4 * 128, ssb, ns);
      sd_kernel<<<cdiv((long)nd * 128, 256), 256, 0, stream>>>(
          cur[dt], wdb + (long)r * 4 * dim, sdb, nd, dim);
      // segment softmax + scatter
      fill_kernel<<<cdiv((long)nd * 4, 256), 256, 0, stream>>>(mb, (long)nd * 4, 0.f);
      fill_kernel<<<cdiv((long)nd * 4, 256), 256, 0, stream>>>(denb, (long)nd * 4, 0.f);
      edge_pass1<<<cdiv((long)NEDGE * 4, 256), 256, 0, stream>>>(
          srcp, dstp, ssb, sdb, eb, (unsigned*)mb, NEDGE);
      edge_pass2<<<cdiv((long)NEDGE * 4, 256), 256, 0, stream>>>(
          dstp, eb, (const unsigned*)mb, denb, NEDGE);
      edge_scatter<<<2048, 256, 0, stream>>>(srcp, dstp, eb, denb, hs, acc[dt], NEDGE);
    }

    // BatchNorm on tx (per-relation bias cancels under BN), relu, residual
    fill_kernel<<<4, 256, 0, stream>>>(stats, 1024, 0.f);
    bn_stats<<<cdiv(N_TX, 128), 256, 0, stream>>>(acc[0], stats, N_TX);
    bn_finalize<<<2, 256, 0, stream>>>(stats, bn_g + (long)i * 512, bn_b + (long)i * 512,
                                       scale, shift, 1.f / (float)N_TX);
    apply_bn<<<cdiv((long)N_TX * 512, 256), 256, 0, stream>>>(
        acc[0], cur[0], scale, shift, (long)N_TX * 512, i > 0);

    bsum_kernel<<<2, 256, 0, stream>>>(att_b + (long)i * 6 * 512, bsacc, bsmer);
    apply_plain<<<cdiv((long)N_ACC * 512, 256), 256, 0, stream>>>(
        acc[1], cur[1], bsacc, (long)N_ACC * 512, i > 0);
    apply_plain<<<cdiv((long)N_MER * 512, 256), 256, 0, stream>>>(
        acc[2], cur[2], bsmer, (long)N_MER * 512, i > 0);

    p ^= 1;  // transformed-in-place accumulation buffers become the new h
  }

  // ---- head: hidden = relu(h_tx @ Wh1 + bh1); logits = hidden @ Wh2 + bh2
  gemm_wmma_kernel<<<dim3(1, cdiv(N_TX, 128)), 256, 0, stream>>>(
      htx[p], Wh1, bh1, hidden, N_TX, 64, 512, 1);
  logits_kernel<<<cdiv((long)N_TX * 32, 256), 256, 0, stream>>>(
      hidden, Wh2, bh2, (float*)d_out, N_TX);
}
